// AdaptiveSTGCNBlock_18433999634864
// MI455X (gfx1250) — compile-verified, matrix operand order checked by reading, not run on hardware
//
#include <hip/hip_runtime.h>
#include <hip/hip_bf16.h>

// ---------------------------------------------------------------------------
// AdaptiveSTGCNBlock on MI455X (gfx1250): all stages as BF16 WMMA GEMMs,
// with CDNA5 async global->LDS staging (ASYNCcnt) for all bf16 tiles.
// B=16, N=128, C_IN=64, C_OUT=128, T=256, K=9, PAD=4
// ---------------------------------------------------------------------------

#define B_   16
#define N_   128
#define CI_  64
#define CO_  128
#define T_   256
#define KW_  9
#define PAD_ 4
#define TT_  64          // t-tile per workgroup in temporal stages
#define SLABT (TT_ + 8)  // 72 rows with +-4 halo

typedef __attribute__((ext_vector_type(16))) __bf16 v16bf;
typedef __attribute__((ext_vector_type(8)))  float  v8f;

union FragU { v16bf v; unsigned u[8]; };

__device__ __forceinline__ unsigned short f2bf(float x) {
  unsigned u = __builtin_bit_cast(unsigned, x);
  unsigned r = (u + 0x7FFFu + ((u >> 16) & 1u)) >> 16;
  return (unsigned short)r;
}

// --- CDNA5 async global->LDS copy (16B per lane), tracked with ASYNCcnt ----
// LDS destination operand is the wave-relative LDS byte offset; the generic
// pointer to a __shared__ object carries that offset in its low 32 bits
// (see ISA 10.2, LDS aperture).
__device__ __forceinline__ unsigned lds_off(const void* p) {
  return (unsigned)(unsigned long long)p;
}
__device__ __forceinline__ void async_b128(unsigned loff, const void* g) {
  asm volatile("global_load_async_to_lds_b128 %0, %1, off"
               :: "v"(loff), "v"((unsigned long long)g)
               : "memory");
}
__device__ __forceinline__ void wait_async0() {
  asm volatile("s_wait_asynccnt 0x0" ::: "memory");
}

// Load a 16x32 bf16 WMMA fragment (A: rows=M, cols=K / B stored col-major:
// rows=N, cols=K) from a row-major [row][strideHalfs] bf16 array viewed as
// u32. The 8 u32s form two contiguous 16B runs -> compiler emits two
// ds_load_b128 (or global_load_b128) per fragment.
__device__ __forceinline__ v16bf load_frag(const unsigned* src32, int rowBase,
                                           int strideHalfs, int kBase, int lane) {
  const int row  = rowBase + (lane & 15);
  const int koff = (lane >> 4) << 3;                  // 0 or 8
  const int base = row * strideHalfs + kBase + koff;  // u16 units, even
  FragU f;
#pragma unroll
  for (int j = 0; j < 8; ++j) {
    const int kk = ((j & 3) << 1) + ((j >> 2) << 4);
    f.u[j] = src32[(base + kk) >> 1];
  }
  return f.v;
}

__device__ __forceinline__ v8f wmma_bf16(v16bf a, v16bf b, v8f c) {
#if defined(__HIP_DEVICE_COMPILE__)
  return __builtin_amdgcn_wmma_f32_16x16x32_bf16(
      /*neg_a=*/false, a, /*neg_b=*/false, b,
      /*c_mod=*/(short)0, c, /*reuse_a=*/false, /*reuse_b=*/false);
#else
  return c;
#endif
}

__device__ __forceinline__ v8f vzero8() {
  v8f z = {0.f, 0.f, 0.f, 0.f, 0.f, 0.f, 0.f, 0.f};
  return z;
}

// ---------------------------------------------------------------------------
// K0: convert weights + adjacency to bf16 with GEMM-friendly layouts.
//   w1b  [k][co][ci]   (9 x 128 x 64)
//   w2b  [k][co][ci]   (9 x 128 x 128)
//   wspT [o][c]        (128 x 128)   (transposed: w_sp is [c][o])
//   wresb[co][ci]      (128 x 64)
//   adjb [b][n][m]     (16 x 128 x 128)
// ---------------------------------------------------------------------------
__global__ __launch_bounds__(256) void k_prep(
    const float* __restrict__ w_t1, const float* __restrict__ w_sp,
    const float* __restrict__ w_t2, const float* __restrict__ w_res,
    const float* __restrict__ adj,
    unsigned short* __restrict__ w1b, unsigned short* __restrict__ wspT,
    unsigned short* __restrict__ w2b, unsigned short* __restrict__ wresb,
    unsigned short* __restrict__ adjb) {
  int i = blockIdx.x * 256 + threadIdx.x;
  const int n1 = KW_ * CO_ * CI_;      // 73728
  const int n2 = KW_ * CO_ * CO_;      // 147456
  const int n3 = CO_ * CO_;            // 16384
  const int n4 = CO_ * CI_;            // 8192
  const int n5 = B_ * N_ * N_;         // 262144
  if (i < n1) {
    int k = i / (CO_ * CI_), r = i % (CO_ * CI_);
    int co = r >> 6, ci = r & 63;
    w1b[i] = f2bf(w_t1[(co * CI_ + ci) * KW_ + k]);
    return;
  }
  i -= n1;
  if (i < n2) {
    int k = i / (CO_ * CO_), r = i % (CO_ * CO_);
    int co = r >> 7, ci = r & 127;
    w2b[i] = f2bf(w_t2[(co * CO_ + ci) * KW_ + k]);
    return;
  }
  i -= n2;
  if (i < n3) {
    int o = i >> 7, c = i & 127;
    wspT[i] = f2bf(w_sp[c * CO_ + o]);
    return;
  }
  i -= n3;
  if (i < n4) { wresb[i] = f2bf(w_res[i]); return; }
  i -= n4;
  if (i < n5) { adjb[i] = f2bf(adj[i]); }
}

// ---------------------------------------------------------------------------
// K1: temporal conv1 (K=9, CI=64 -> CO=128) + bias + ReLU.
// One WG per (b, n, t-tile). Output bf16 in (b, t, n, c) layout.
// Weight slices are async-copied to LDS per tap.
// ---------------------------------------------------------------------------
__global__ __launch_bounds__(256) void k_temporal1(
    const float* __restrict__ x, const unsigned short* __restrict__ w1b,
    const float* __restrict__ b_t1, unsigned short* __restrict__ hbuf) {
  __shared__ unsigned short slab[SLABT * CI_];  // [tt][ci], bf16
  __shared__ unsigned short w1s[CO_ * CI_];     // [co][ci], per-tap slice

  const int tid  = threadIdx.x;
  const int lane = tid & 31;
  const int w    = tid >> 5;
  const int blk  = blockIdx.x;
  const int ttile = blk & 3;
  const int seq   = blk >> 2;       // b*N + n
  const int b     = seq >> 7;
  const int n     = seq & 127;
  const int t0    = ttile * TT_;

  // stage x slab with halo, f32 -> bf16 (VALU conversion path)
  const float* xs = x + (long)seq * CI_ * T_;
  for (int e = tid; e < SLABT * CI_; e += 256) {
    int ci = e / SLABT, tt = e % SLABT;
    int t = t0 + tt - PAD_;
    float v = (t >= 0 && t < T_) ? xs[ci * T_ + t] : 0.0f;
    slab[tt * CI_ + ci] = f2bf(v);
  }

  const int tRow   = (w & 3) * 16;   // wave's 16 t-rows
  const int coBase = (w >> 2) * 64;  // wave's 64 co columns
  v8f acc[4];
#pragma unroll
  for (int j = 0; j < 4; ++j) acc[j] = vzero8();

  const unsigned* slab32 = (const unsigned*)slab;
  const unsigned* w1s32  = (const unsigned*)w1s;
  const unsigned wbase   = lds_off(w1s);

  for (int k = 0; k < KW_; ++k) {
    __syncthreads();  // all waves done reading previous tap's w1s
    {
      const char* gsrc = (const char*)(w1b + k * (CO_ * CI_));
#pragma unroll
      for (int i = 0; i < 4; ++i) {      // 16 KB = 1024 x 16B chunks
        int c = tid + i * 256;
        async_b128(wbase + c * 16, gsrc + c * 16);
      }
    }
    wait_async0();
    __syncthreads();
#pragma unroll
    for (int cb = 0; cb < CI_; cb += 32) {
      v16bf a = load_frag(slab32, tRow + k, CI_, cb, lane);
#pragma unroll
      for (int j = 0; j < 4; ++j) {
        v16bf bf = load_frag(w1s32, coBase + j * 16, CI_, cb, lane);
        acc[j] = wmma_bf16(a, bf, acc[j]);
      }
    }
  }

  // epilogue: bias + ReLU, bf16 store to hbuf[b][t][n][co]
  const int col = lane & 15;
  const int mhi = (lane >> 4) << 3;
#pragma unroll
  for (int j = 0; j < 4; ++j) {
    const int co = coBase + j * 16 + col;
    const float bias = b_t1[co];
#pragma unroll
    for (int r = 0; r < 8; ++r) {
      const int t = t0 + tRow + r + mhi;
      float v = acc[j][r] + bias;
      v = v > 0.f ? v : 0.f;
      hbuf[((long)(b * T_ + t) * N_ + n) * CO_ + co] = f2bf(v);
    }
  }
}

// ---------------------------------------------------------------------------
// K2: spatial graph conv. One WG per (b, t):
//   S = H (128n x 128c) * Wsp ; Z = Adj_b * S ; +b_sp ; ReLU.
// H/Wsp/Adj tiles async-copied to LDS; S^T kept in LDS (aliased over H).
// In-place update of hbuf tile (WG reads its whole tile into LDS first).
// ---------------------------------------------------------------------------
__global__ __launch_bounds__(256) void k_spatial(
    unsigned short* __restrict__ hbuf, const unsigned short* __restrict__ wspT,
    const unsigned short* __restrict__ adjb, const float* __restrict__ b_sp) {
  __shared__ unsigned short bufA[N_ * CO_];  // phase1: H [n][c]; phase2: S^T [o][m]
  __shared__ unsigned short bufB[N_ * CO_];  // phase1: WspT [o][c]; phase2: Adj [n][m]

  const int tid  = threadIdx.x;
  const int lane = tid & 31;
  const int w    = tid >> 5;
  const int blk  = blockIdx.x;
  const int b    = blk >> 8;
  const int t    = blk & 255;

  const unsigned abase = lds_off(bufA);
  const unsigned bbase = lds_off(bufB);

  // async-stage H tile (32 KB contiguous) and WspT (32 KB)
  {
    const char* hsrc = (const char*)(hbuf + (long)(b * T_ + t) * N_ * CO_);
    const char* wsrc = (const char*)wspT;
#pragma unroll
    for (int i = 0; i < 8; ++i) {        // 2048 x 16B chunks each
      int c = tid + i * 256;
      async_b128(abase + c * 16, hsrc + c * 16);
      async_b128(bbase + c * 16, wsrc + c * 16);
    }
  }
  wait_async0();
  __syncthreads();

  const int nBase = w * 16;
  const int col   = lane & 15;
  const int mhi   = (lane >> 4) << 3;

  // phase 1: S[n][o] = sum_c H[n][c] * Wsp[c][o]
  v8f acc[8];
#pragma unroll
  for (int j = 0; j < 8; ++j) acc[j] = vzero8();
#pragma unroll
  for (int cb = 0; cb < CO_; cb += 32) {
    v16bf a = load_frag((const unsigned*)bufA, nBase, CO_, cb, lane);
#pragma unroll
    for (int j = 0; j < 8; ++j) {
      v16bf bf = load_frag((const unsigned*)bufB, j * 16, CO_, cb, lane);
      acc[j] = wmma_bf16(a, bf, acc[j]);
    }
  }
  __syncthreads();  // all phase-1 LDS reads complete

  // write S^T into bufA ([o][m]); async-stage adjacency into bufB ([n][m])
#pragma unroll
  for (int j = 0; j < 8; ++j) {
    const int o = j * 16 + col;
#pragma unroll
    for (int r = 0; r < 8; ++r) {
      const int m = nBase + r + mhi;
      bufA[o * N_ + m] = f2bf(acc[j][r]);
    }
  }
  {
    const char* asrc = (const char*)(adjb + (long)b * N_ * N_);
#pragma unroll
    for (int i = 0; i < 8; ++i) {
      int c = tid + i * 256;
      async_b128(bbase + c * 16, asrc + c * 16);
    }
  }
  wait_async0();
  __syncthreads();

  // phase 2: Z[n][o] = sum_m Adj[n][m] * S[m][o]
  v8f acc2[8];
#pragma unroll
  for (int j = 0; j < 8; ++j) acc2[j] = vzero8();
#pragma unroll
  for (int mb = 0; mb < N_; mb += 32) {
    v16bf a = load_frag((const unsigned*)bufB, nBase, N_, mb, lane);
#pragma unroll
    for (int j = 0; j < 8; ++j) {
      v16bf bf = load_frag((const unsigned*)bufA, j * 16, N_, mb, lane);
      acc2[j] = wmma_bf16(a, bf, acc2[j]);
    }
  }

  // epilogue: bias + ReLU, in-place bf16 store to hbuf[b][t][n][o]
#pragma unroll
  for (int j = 0; j < 8; ++j) {
    const int o = j * 16 + col;
    const float bias = b_sp[o];
#pragma unroll
    for (int r = 0; r < 8; ++r) {
      const int m = nBase + r + mhi;
      float v = acc2[j][r] + bias;
      v = v > 0.f ? v : 0.f;
      hbuf[((long)(b * T_ + t) * N_ + m) * CO_ + o] = f2bf(v);
    }
  }
}

// ---------------------------------------------------------------------------
// K3: temporal conv2 (K=9, CO->CO) + residual 1x1 (CI->CO) + biases.
// h2 slab + per-tap weight slices async-copied to LDS.
// One WG per (b, n, t-tile). Output f32 (b, n, c, t) via LDS transpose tile.
// ---------------------------------------------------------------------------
__global__ __launch_bounds__(256) void k_temporal2(
    const unsigned short* __restrict__ hbuf, const float* __restrict__ x,
    const unsigned short* __restrict__ w2b, const unsigned short* __restrict__ wresb,
    const float* __restrict__ b_t2, const float* __restrict__ b_res,
    float* __restrict__ out) {
  __shared__ unsigned short slab[SLABT * CO_];  // h2 slab [tt][ci]  (18 KB)
  __shared__ unsigned short xslab[TT_ * CI_];   // x slab  [tl][ci]  ( 8 KB)
  __shared__ unsigned short w2s[CO_ * CO_];     // [co][ci] per tap  (32 KB)
  float* otile = (float*)w2s;                   // aliased: [co][tl] f32 (32 KB)

  const int tid  = threadIdx.x;
  const int lane = tid & 31;
  const int w    = tid >> 5;
  const int blk  = blockIdx.x;
  const int ttile = blk & 3;
  const int seq   = blk >> 2;
  const int b     = seq >> 7;
  const int n     = seq & 127;
  const int t0    = ttile * TT_;

  // async-stage h2 slab with halo: 72 rows x 256B; OOB rows zero-filled in LDS
  {
    const unsigned sbase = lds_off(slab);
#pragma unroll
    for (int i = 0; i < 5; ++i) {            // 72*16 = 1152 x 16B chunks
      int e = tid + i * 256;
      if (e < SLABT * 16) {
        int tt = e >> 4, part = e & 15;
        int t = t0 + tt - PAD_;
        if (t >= 0 && t < T_) {
          const char* src =
              (const char*)(hbuf + ((long)(b * T_ + t) * N_ + n) * CO_) + part * 16;
          async_b128(sbase + e * 16, src);
        } else {
          uint4 z = {0u, 0u, 0u, 0u};
          *(uint4*)((char*)slab + e * 16) = z;
        }
      }
    }
  }
  // stage x slab (f32 -> bf16), no halo needed for 1x1 residual
  {
    const float* xs = x + (long)seq * CI_ * T_;
    for (int e = tid; e < TT_ * CI_; e += 256) {
      int ci = e >> 6, tl = e & 63;
      xslab[tl * CI_ + ci] = f2bf(xs[ci * T_ + t0 + tl]);
    }
  }

  const int tRow   = (w & 3) * 16;
  const int coBase = (w >> 2) * 64;
  const int col    = lane & 15;
  const int mhi    = (lane >> 4) << 3;

  v8f acc[4];
#pragma unroll
  for (int j = 0; j < 4; ++j) acc[j] = vzero8();

  const unsigned* slab32  = (const unsigned*)slab;
  const unsigned* xslab32 = (const unsigned*)xslab;
  const unsigned* w2s32   = (const unsigned*)w2s;
  const unsigned  w2base  = lds_off(w2s);

  // main 9-tap conv: K = 9 * 128
  for (int k = 0; k < KW_; ++k) {
    __syncthreads();  // all waves done reading previous tap's w2s
    {
      const char* gsrc = (const char*)(w2b + k * (CO_ * CO_));
#pragma unroll
      for (int i = 0; i < 8; ++i) {      // 32 KB = 2048 x 16B chunks
        int c = tid + i * 256;
        async_b128(w2base + c * 16, gsrc + c * 16);
      }
    }
    wait_async0();                        // also covers slab on first tap
    __syncthreads();
#pragma unroll
    for (int cb = 0; cb < CO_; cb += 32) {
      v16bf a = load_frag(slab32, tRow + k, CO_, cb, lane);
#pragma unroll
      for (int j = 0; j < 4; ++j) {
        v16bf bf = load_frag(w2s32, coBase + j * 16, CO_, cb, lane);
        acc[j] = wmma_bf16(a, bf, acc[j]);
      }
    }
  }

  // residual GEMM: out += X[t][ci] * Wres[ci][co]; B frags straight from global
  {
    const unsigned* wres32 = (const unsigned*)wresb;
#pragma unroll
    for (int cb = 0; cb < CI_; cb += 32) {
      v16bf a = load_frag(xslab32, tRow, CI_, cb, lane);
#pragma unroll
      for (int j = 0; j < 4; ++j) {
        v16bf bf = load_frag(wres32, coBase + j * 16, CI_, cb, lane);
        acc[j] = wmma_bf16(a, bf, acc[j]);
      }
    }
  }

  __syncthreads();  // all w2s reads done before otile aliases it

  // dump accumulators (+ biases) into otile[co][tl]
#pragma unroll
  for (int j = 0; j < 4; ++j) {
    const int co = coBase + j * 16 + col;
    const float bias = b_t2[co] + b_res[co];
#pragma unroll
    for (int r = 0; r < 8; ++r) {
      const int tl = tRow + r + mhi;
      otile[co * TT_ + tl] = acc[j][r] + bias;
    }
  }
  __syncthreads();

  // coalesced f32 output: out[b][n][co][t0 + tl]
  {
    float* obase = out + ((long)seq * CO_) * T_ + t0;
    for (int e = tid; e < CO_ * (TT_ / 4); e += 256) {
      const int co = e >> 4;
      const int tq = (e & 15) << 2;
      const float4 v = *(const float4*)&otile[co * TT_ + tq];
      *(float4*)&obase[co * T_ + tq] = v;
    }
  }
}

// ---------------------------------------------------------------------------
// launch
// ---------------------------------------------------------------------------
extern "C" void kernel_launch(void* const* d_in, const int* in_sizes, int n_in,
                              void* d_out, int out_size, void* d_ws, size_t ws_size,
                              hipStream_t stream) {
  (void)in_sizes; (void)n_in; (void)out_size; (void)ws_size;

  const float* x     = (const float*)d_in[0];
  const float* adj   = (const float*)d_in[1];
  const float* w_t1  = (const float*)d_in[2];
  const float* b_t1  = (const float*)d_in[3];
  const float* w_sp  = (const float*)d_in[4];
  const float* b_sp  = (const float*)d_in[5];
  const float* w_t2  = (const float*)d_in[6];
  const float* b_t2  = (const float*)d_in[7];
  const float* w_res = (const float*)d_in[8];
  const float* b_res = (const float*)d_in[9];
  float* out = (float*)d_out;

  unsigned short* us    = (unsigned short*)d_ws;
  unsigned short* hbuf  = us;                        // 67,108,864 bf16 (128 MB)
  unsigned short* w1b   = us + 67108864ull;          // 9*128*64
  unsigned short* w2b   = w1b + 73728;               // 9*128*128
  unsigned short* wspT  = w2b + 147456;              // 128*128
  unsigned short* wresb = wspT + 16384;              // 128*64
  unsigned short* adjb  = wresb + 8192;              // 16*128*128

  k_prep<<<1984, 256, 0, stream>>>(w_t1, w_sp, w_t2, w_res, adj,
                                   w1b, wspT, w2b, wresb, adjb);
  k_temporal1<<<B_ * N_ * (T_ / TT_), 256, 0, stream>>>(x, w1b, b_t1, hbuf);
  k_spatial<<<B_ * T_, 256, 0, stream>>>(hbuf, wspT, adjb, b_sp);
  k_temporal2<<<B_ * N_ * (T_ / TT_), 256, 0, stream>>>(hbuf, x, w2b, wresb,
                                                        b_t2, b_res, out);
}
